// BipartiteImputer_89361089560737
// MI455X (gfx1250) — matrix operand order repaired and checked by reading, not compiled, not run on hardware
//
#include <hip/hip_runtime.h>
#include <hip/hip_bf16.h>
#include <math.h>

// ---------------- problem constants ----------------
#define NS      4096
#define FDIM    128
#define EDIM    128
#define HDIM    256
#define NODE_IN 130
#define EDGE_IN 261
#define KP_EDGE 288   // EDGE_IN padded to multiple of 32
#define DIN_NODE 386  // NODE_IN + HDIM
#define KP_NODE 416   // padded to 32
#define NP_NODE 144   // NODE_IN padded to 16
#define SFS     144   // s_feat / f_feat row stride (f32)

typedef __bf16 bf16_t;
typedef __attribute__((ext_vector_type(16))) __bf16 bf16x16;
typedef __attribute__((ext_vector_type(8)))  float   f32x8;

// ---------------- device helpers ----------------
__device__ __forceinline__ float wave_sum(float v) {
#pragma unroll
  for (int o = 16; o > 0; o >>= 1) v += __shfl_xor(v, o, 32);
  return v;
}
__device__ __forceinline__ float gelu_f(float x) {
  return 0.5f * x * (1.0f + erff(x * 0.70710678118654752f));
}
__device__ __forceinline__ float n2n(float x) {
  if (isnan(x)) return 0.0f;
  if (isinf(x)) return x > 0.f ? 3.4028235e38f : -3.4028235e38f;
  return x;
}

// Generic block-cooperative GEMM: C[rows x npad] (f32, LDS) = A[rows x kpad] (bf16, LDS)
//  @ Bpk (bf16, packed fragment-major in global).
// ACC=false: accumulator initialized from bias[col] (bias must be npad floats, padded).
// ACC=true : accumulator initialized from existing C (bias ignored).
// rows, npad multiples of 16; kpad multiple of 32.
template <bool ACC>
__device__ __forceinline__ void gemm_tiles(const bf16_t* A, int kpad,
                                           const bf16_t* __restrict__ Bpk, int npad,
                                           float* C, int ldc,
                                           const float* __restrict__ bias,
                                           int rows, int tid, int nthr) {
  const int lane = tid & 31;
  const int wv   = tid >> 5;
  const int nw   = nthr >> 5;
  const int mt = rows >> 4, nt = npad >> 4;
  const int kbc = kpad >> 5;
  const int m = lane & 15, h = lane >> 4;
  for (int t = wv; t < mt * nt; t += nw) {
    const int mi = t / nt, ni = t - mi * nt;
    const int col = ni * 16 + m;
    f32x8 c;
    if (ACC) {
#pragma unroll
      for (int r = 0; r < 8; ++r)
        c[r] = C[(size_t)(mi * 16 + r + 8 * h) * ldc + col];
    } else {
      const float binit = bias[col];   // single load, broadcast to all 8 slots
#pragma unroll
      for (int r = 0; r < 8; ++r) c[r] = binit;
    }
    for (int kb = 0; kb < kbc; ++kb) {
      // A fragment: 16-bit A 16x32 layout, K = 16*(e/8) + 8*h + e%8
      const bf16_t* ar = A + (size_t)(mi * 16 + m) * kpad + kb * 32;
      bf16x16 a;
#pragma unroll
      for (int e = 0; e < 8; ++e) a[e] = ar[8 * h + e];
#pragma unroll
      for (int e = 0; e < 8; ++e) a[8 + e] = ar[16 + 8 * h + e];
      // B fragment: packed contiguous 16 bf16 per lane (32B -> 2x global_load_b128)
      const bf16_t* bp = Bpk + (((size_t)kb * nt + ni) * 32 + lane) * 16;
      bf16x16 b;
#pragma unroll
      for (int e = 0; e < 16; ++e) b[e] = bp[e];
      c = __builtin_amdgcn_wmma_f32_16x16x32_bf16(false, a, false, b,
                                                  (short)0, c, false, false);
    }
#pragma unroll
    for (int r = 0; r < 8; ++r)
      C[(size_t)(mi * 16 + r + 8 * h) * ldc + col] = c[r];
  }
}

// Row-wise LayerNorm(+GELU), C f32 [rows x ldc] -> bf16 out [rows x ldo] (hid == ldo)
__device__ __forceinline__ void ln_gelu_rows(const float* C, int ldc, int hid,
                                             int rows, const float* __restrict__ g,
                                             const float* __restrict__ be, bf16_t* out,
                                             int ldo, int tid, int nthr) {
  const int lane = tid & 31, wv = tid >> 5, nw = nthr >> 5;
  for (int r = wv; r < rows; r += nw) {
    const float* row = C + (size_t)r * ldc;
    float s = 0.f;
    for (int c = lane; c < hid; c += 32) s += row[c];
    const float mean = wave_sum(s) / (float)hid;
    float v = 0.f;
    for (int c = lane; c < hid; c += 32) { float d = row[c] - mean; v += d * d; }
    const float inv = rsqrtf(wave_sum(v) / (float)hid + 1e-5f);
    for (int c = lane; c < hid; c += 32) {
      float y = (row[c] - mean) * inv * g[c] + be[c];
      out[(size_t)r * ldo + c] = (bf16_t)gelu_f(y);
    }
  }
}

// Assemble 128 edge rows [s_feat(n) | f_feat(f) | w] zero-padded to KP_EDGE (bf16)
__device__ __forceinline__ void assemble_edge(bf16_t* eA, const float* __restrict__ sfeat,
                                              const float* __restrict__ ffeat,
                                              const float* __restrict__ wrow, bool fix,
                                              int n, int tid, int nthr) {
  const float* srow = sfeat + (size_t)n * SFS;
  for (int i = tid; i < FDIM * KP_EDGE; i += nthr) {
    const int f = i / KP_EDGE, c = i - f * KP_EDGE;
    float v;
    if (c < NODE_IN)            v = srow[c];
    else if (c < 2 * NODE_IN)   v = ffeat[(size_t)f * SFS + (c - NODE_IN)];
    else if (c == 2 * NODE_IN) { v = wrow[f]; if (fix) v = n2n(v); }
    else                        v = 0.f;
    eA[i] = (bf16_t)v;
  }
}

// ---------------- kernels ----------------

// pack f32 weight [din x dout] -> bf16 WMMA-B fragment order [kb][nb][lane][16]
__global__ void pack_b_kernel(const float* __restrict__ src, bf16_t* __restrict__ dst,
                              int din, int dout, int kpad, int npad) {
  const int idx = blockIdx.x * blockDim.x + threadIdx.x;
  const int nbc = npad >> 4;
  const int total = (kpad >> 5) * nbc * 512;
  if (idx >= total) return;
  const int e = idx & 15, lane = (idx >> 4) & 31, tile = idx >> 9;
  const int kb = tile / nbc, nb = tile - kb * nbc;
  const int nl = lane & 15, h = lane >> 4;
  const int K = kb * 32 + 16 * (e >> 3) + 8 * h + (e & 7);
  const int Nn = nb * 16 + nl;
  float v = (K < din && Nn < dout) ? src[(size_t)K * dout + Nn] : 0.0f;
  dst[idx] = (bf16_t)v;
}

// pad a bias vector: dst[0..npad) = (i < n) ? src[i] : 0
__global__ void pad_bias_kernel(const float* __restrict__ src, float* __restrict__ dst,
                                int n, int npad) {
  int i = blockIdx.x * blockDim.x + threadIdx.x;
  if (i < npad) dst[i] = (i < n) ? src[i] : 0.f;
}

__global__ void zero_kernel(float* p, int n) {
  int i = blockIdx.x * blockDim.x + threadIdx.x;
  if (i < n) p[i] = 0.f;
}

__global__ void winit_kernel(const float* __restrict__ W, float* __restrict__ w0, int n) {
  int i = blockIdx.x * blockDim.x + threadIdx.x;
  if (i < n) w0[i] = n2n(W[i]);
}

__global__ void ffeat_init_kernel(const float* __restrict__ femb, float* __restrict__ ffeat) {
  int i = blockIdx.x * blockDim.x + threadIdx.x;
  if (i >= FDIM * SFS) return;
  int f = i / SFS, c = i - f * SFS;
  float v = (c < EDIM) ? femb[(size_t)f * EDIM + c] : (c == 129 ? 1.f : 0.f);
  ffeat[i] = v;
}

// block = sample n. val MLP + fuse + masked mean over features.
__global__ void encoder_kernel(const float* __restrict__ W, const float* __restrict__ M,
                               const float* __restrict__ fbase,
                               const float* __restrict__ vw1, const float* __restrict__ vb1,
                               const bf16_t* __restrict__ vw2pk, const float* __restrict__ vb2,
                               const bf16_t* __restrict__ fusepk, const float* __restrict__ fuseb,
                               float* __restrict__ mean_feats) {
  extern __shared__ char smem[];
  bf16_t* hbf = (bf16_t*)smem;                    // [128][128] bf16
  float*  val = (float*)(smem + 32768);           // [128][128] f32 (reused as C)
  bf16_t* A2  = (bf16_t*)(smem + 98304);          // [128][256] bf16
  const int n = blockIdx.x, tid = threadIdx.x, nthr = blockDim.x;

  for (int i = tid; i < FDIM * EDIM; i += nthr) {
    int f = i / EDIM, e = i - f * EDIM;
    float x = n2n(W[(size_t)n * FDIM + f]);
    hbf[i] = (bf16_t)gelu_f(x * vw1[e] + vb1[e]);
  }
  __syncthreads();
  gemm_tiles<false>(hbf, EDIM, vw2pk, EDIM, val, EDIM, vb2, FDIM, tid, nthr);
  __syncthreads();
  for (int i = tid; i < FDIM * 2 * EDIM; i += nthr) {
    int f = i / (2 * EDIM), c = i - f * 2 * EDIM;
    float v = (c < EDIM) ? val[(size_t)f * EDIM + c] : fbase[(size_t)f * EDIM + (c - EDIM)];
    A2[i] = (bf16_t)v;
  }
  __syncthreads();
  gemm_tiles<false>(A2, 2 * EDIM, fusepk, EDIM, val, EDIM, fuseb, FDIM, tid, nthr);
  __syncthreads();
  if (tid < EDIM) {
    float cnt = 0.f, s = 0.f;
    for (int f = 0; f < FDIM; ++f) {
      float m = M[(size_t)n * FDIM + f];
      cnt += m;
      s += val[(size_t)f * EDIM + tid] * m;
    }
    if (cnt < 1.f) cnt = 1.f;
    mean_feats[(size_t)n * EDIM + tid] = s / cnt;
  }
}

// block = 16 samples. sp MLP -> s_feat = [s_base, 1, 0] (stride SFS)
__global__ void sp_kernel(const float* __restrict__ mean_feats,
                          const bf16_t* __restrict__ sw1pk, const float* __restrict__ sb1,
                          const float* __restrict__ sg, const float* __restrict__ sbe,
                          const bf16_t* __restrict__ sw2pk, const float* __restrict__ sb2,
                          float* __restrict__ sfeat) {
  extern __shared__ char smem[];
  bf16_t* A = (bf16_t*)smem;               // [16][128]
  float*  C = (float*)(smem + 4096);       // [16][128]
  bf16_t* hb = (bf16_t*)(smem + 12288);    // [16][128]
  const int tid = threadIdx.x, nthr = blockDim.x;
  const int row0 = blockIdx.x * 16;
  for (int i = tid; i < 16 * EDIM; i += nthr)
    A[i] = (bf16_t)mean_feats[(size_t)row0 * EDIM + i];
  __syncthreads();
  gemm_tiles<false>(A, EDIM, sw1pk, EDIM, C, EDIM, sb1, 16, tid, nthr);
  __syncthreads();
  ln_gelu_rows(C, EDIM, EDIM, 16, sg, sbe, hb, EDIM, tid, nthr);
  __syncthreads();
  gemm_tiles<false>(hb, EDIM, sw2pk, EDIM, C, EDIM, sb2, 16, tid, nthr);
  __syncthreads();
  for (int i = tid; i < 16 * SFS; i += nthr) {
    int r = i / SFS, c = i - r * SFS;
    float v = (c < EDIM) ? C[(size_t)r * EDIM + c] : (c == 128 ? 1.f : 0.f);
    sfeat[(size_t)(row0 + r) * SFS + c] = v;
  }
}

// block = sample n: attention logits + in-block axis-1 softmax
__global__ void attn_kernel(const float* __restrict__ sfeat, const float* __restrict__ ffeat,
                            const float* __restrict__ wbuf, const float* __restrict__ M,
                            const bf16_t* __restrict__ aw1pk, const float* __restrict__ ab1,
                            const float* __restrict__ aw2,
                            float* __restrict__ a_out, float* __restrict__ attn_s) {
  extern __shared__ char smem[];
  bf16_t* eA = (bf16_t*)smem;                      // [128][288]
  float*  Cf = (float*)(smem + 73728);             // [128][128]
  float*  ar = (float*)(smem + 73728 + 65536);     // [128]
  float*  red = ar + 128;                          // [128]
  const int n = blockIdx.x, tid = threadIdx.x, nthr = blockDim.x;
  const int lane = tid & 31, wv = tid >> 5, nw = nthr >> 5;

  assemble_edge(eA, sfeat, ffeat, wbuf + (size_t)n * FDIM, false, n, tid, nthr);
  __syncthreads();
  gemm_tiles<false>(eA, KP_EDGE, aw1pk, EDIM, Cf, EDIM, ab1, FDIM, tid, nthr);
  __syncthreads();
  for (int f = wv; f < FDIM; f += nw) {
    float s = 0.f;
    for (int c = lane; c < EDIM; c += 32) s += tanhf(Cf[(size_t)f * EDIM + c]) * aw2[c];
    s = wave_sum(s);
    if (lane == 0) {
      float av = (M[(size_t)n * FDIM + f] == 0.f) ? -1e9f : s;
      ar[f] = av;
      a_out[(size_t)n * FDIM + f] = av;
    }
  }
  __syncthreads();
  if (tid < 128) red[tid] = ar[tid];
  __syncthreads();
  for (int s = 64; s > 0; s >>= 1) { if (tid < s) red[tid] = fmaxf(red[tid], red[tid + s]); __syncthreads(); }
  const float mx = red[0];
  __syncthreads();
  if (tid < 128) { float e = expf(ar[tid] - mx); ar[tid] = e; red[tid] = e; }
  __syncthreads();
  for (int s = 64; s > 0; s >>= 1) { if (tid < s) red[tid] += red[tid + s]; __syncthreads(); }
  if (tid < 128) attn_s[(size_t)n * FDIM + tid] = ar[tid] / red[0];
}

// block = feature f: softmax over samples (axis 0)
__global__ void softmax_axis0_kernel(const float* __restrict__ a, float* __restrict__ attn_f) {
  __shared__ float red[256];
  const int f = blockIdx.x, t = threadIdx.x;
  float mx = -3.4e38f;
  for (int nI = t; nI < NS; nI += 256) mx = fmaxf(mx, a[(size_t)nI * FDIM + f]);
  red[t] = mx; __syncthreads();
  for (int s = 128; s > 0; s >>= 1) { if (t < s) red[t] = fmaxf(red[t], red[t + s]); __syncthreads(); }
  mx = red[0]; __syncthreads();
  float sm = 0.f;
  for (int nI = t; nI < NS; nI += 256) sm += expf(a[(size_t)nI * FDIM + f] - mx);
  red[t] = sm; __syncthreads();
  for (int s = 128; s > 0; s >>= 1) { if (t < s) red[t] += red[t + s]; __syncthreads(); }
  const float inv = 1.0f / red[0];
  for (int nI = t; nI < NS; nI += 256)
    attn_f[(size_t)nI * FDIM + f] = expf(a[(size_t)nI * FDIM + f] - mx) * inv;
}

// block = sample n: edge ResMLP messages, attention aggregation, ew ResMLP w-update
__global__ void edge_layer_kernel(const float* __restrict__ sfeat, const float* __restrict__ ffeat,
                                  const float* __restrict__ wbuf,
                                  const float* __restrict__ attn_s, const float* __restrict__ attn_f,
                                  const bf16_t* __restrict__ ew1pk, const float* __restrict__ eb1,
                                  const float* __restrict__ eg, const float* __restrict__ ebe,
                                  const bf16_t* __restrict__ ew2pk, const float* __restrict__ eb2,
                                  const bf16_t* __restrict__ ewppk, const float* __restrict__ ebp,
                                  const bf16_t* __restrict__ ww1pk, const float* __restrict__ wb1,
                                  const float* __restrict__ wg, const float* __restrict__ wbe,
                                  const float* __restrict__ ww2, const float* __restrict__ wb2,
                                  const float* __restrict__ wwp, const float* __restrict__ wbp,
                                  float* __restrict__ agg_s, float* __restrict__ agg_f,
                                  float* __restrict__ w_out) {
  extern __shared__ char smem[];
  bf16_t* eA  = (bf16_t*)smem;                    // [128][288]
  float*  Cf  = (float*)(smem + 73728);           // [128][256]
  bf16_t* hbf = (bf16_t*)(smem + 73728 + 131072); // [128][256]
  const int n = blockIdx.x, tid = threadIdx.x, nthr = blockDim.x;
  const int lane = tid & 31, wv = tid >> 5, nw = nthr >> 5;

  assemble_edge(eA, sfeat, ffeat, wbuf + (size_t)n * FDIM, false, n, tid, nthr);
  __syncthreads();
  // edge ResMLP
  gemm_tiles<false>(eA, KP_EDGE, ew1pk, HDIM, Cf, HDIM, eb1, FDIM, tid, nthr);
  __syncthreads();
  ln_gelu_rows(Cf, HDIM, HDIM, FDIM, eg, ebe, hbf, HDIM, tid, nthr);
  __syncthreads();
  gemm_tiles<false>(hbf, HDIM, ew2pk, HDIM, Cf, HDIM, eb2, FDIM, tid, nthr);
  __syncthreads();
  gemm_tiles<true>(eA, KP_EDGE, ewppk, HDIM, Cf, HDIM, nullptr, FDIM, tid, nthr);
  __syncthreads();
  // aggregation: agg_s[n,:] and atomics into agg_f[f,:] (proj bias ebp added here)
  if (tid < HDIM) {
    const int c = tid;
    const float eb = ebp[c];
    float asum = 0.f;
    for (int f = 0; f < FDIM; ++f) {
      float em = Cf[(size_t)f * HDIM + c] + eb;
      asum += attn_s[(size_t)n * FDIM + f] * em;
      atomicAdd(&agg_f[(size_t)f * HDIM + c], attn_f[(size_t)n * FDIM + f] * em);
    }
    agg_s[(size_t)n * HDIM + c] = asum;
  }
  __syncthreads();
  // ew ResMLP (scalar output) -> w update
  gemm_tiles<false>(eA, KP_EDGE, ww1pk, HDIM, Cf, HDIM, wb1, FDIM, tid, nthr);
  __syncthreads();
  ln_gelu_rows(Cf, HDIM, HDIM, FDIM, wg, wbe, hbf, HDIM, tid, nthr);
  __syncthreads();
  for (int f = wv; f < FDIM; f += nw) {
    float d = 0.f, dp = 0.f;
    for (int c = lane; c < HDIM; c += 32) d += (float)hbf[(size_t)f * HDIM + c] * ww2[c];
    for (int c = lane; c < EDGE_IN; c += 32) dp += (float)eA[(size_t)f * KP_EDGE + c] * wwp[c];
    d = wave_sum(d); dp = wave_sum(dp);
    if (lane == 0)
      w_out[(size_t)n * FDIM + f] =
          wbuf[(size_t)n * FDIM + f] + (d + wb2[0]) + (dp + wbp[0]);
  }
}

// block = 16 node rows: ResMLP on [feat | agg] -> out feat (NODE_IN cols, SFS stride)
// b2p must be the NP_NODE-padded second-layer bias.
__global__ void node_resmlp_kernel(const float* __restrict__ feat, const float* __restrict__ agg,
                                   const bf16_t* __restrict__ w1pk, const float* __restrict__ b1,
                                   const float* __restrict__ g, const float* __restrict__ be,
                                   const bf16_t* __restrict__ w2pk, const float* __restrict__ b2p,
                                   const bf16_t* __restrict__ wppk, const float* __restrict__ bp,
                                   float* __restrict__ outfeat) {
  extern __shared__ char smem[];
  bf16_t* A   = (bf16_t*)smem;                     // [16][416]
  float*  C   = (float*)(smem + 13312);            // [16][256]
  bf16_t* hb  = (bf16_t*)(smem + 13312 + 16384);   // [16][256]
  float*  Co  = (float*)(smem + 13312 + 16384 + 8192); // [16][144]
  const int tid = threadIdx.x, nthr = blockDim.x;
  const int row0 = blockIdx.x * 16;
  for (int i = tid; i < 16 * KP_NODE; i += nthr) {
    int r = i / KP_NODE, c = i - r * KP_NODE;
    float v;
    if (c < NODE_IN)        v = feat[(size_t)(row0 + r) * SFS + c];
    else if (c < DIN_NODE)  v = agg[(size_t)(row0 + r) * HDIM + (c - NODE_IN)];
    else                    v = 0.f;
    A[i] = (bf16_t)v;
  }
  __syncthreads();
  gemm_tiles<false>(A, KP_NODE, w1pk, HDIM, C, HDIM, b1, 16, tid, nthr);
  __syncthreads();
  ln_gelu_rows(C, HDIM, HDIM, 16, g, be, hb, HDIM, tid, nthr);
  __syncthreads();
  gemm_tiles<false>(hb, HDIM, w2pk, NP_NODE, Co, NP_NODE, b2p, 16, tid, nthr);
  __syncthreads();
  gemm_tiles<true>(A, KP_NODE, wppk, NP_NODE, Co, NP_NODE, nullptr, 16, tid, nthr);
  __syncthreads();
  for (int i = tid; i < 16 * SFS; i += nthr) {
    int r = i / SFS, c = i - r * SFS;
    float v = (c < NODE_IN) ? (Co[(size_t)r * NP_NODE + c] + bp[c]) : 0.f;
    outfeat[(size_t)(row0 + r) * SFS + c] = v;
  }
}

// block = sample n: final imputation head, out = W + corr
__global__ void imp_kernel(const float* __restrict__ W,
                           const float* __restrict__ sfeat, const float* __restrict__ ffeat,
                           const bf16_t* __restrict__ iw1pk, const float* __restrict__ ib1,
                           const float* __restrict__ iw2, const float* __restrict__ ib2,
                           float* __restrict__ out) {
  extern __shared__ char smem[];
  bf16_t* eA = (bf16_t*)smem;            // [128][288]
  float*  Cf = (float*)(smem + 73728);   // [128][256]
  const int n = blockIdx.x, tid = threadIdx.x, nthr = blockDim.x;
  const int lane = tid & 31, wv = tid >> 5, nw = nthr >> 5;
  assemble_edge(eA, sfeat, ffeat, W + (size_t)n * FDIM, true, n, tid, nthr);
  __syncthreads();
  gemm_tiles<false>(eA, KP_EDGE, iw1pk, HDIM, Cf, HDIM, ib1, FDIM, tid, nthr);
  __syncthreads();
  for (int f = wv; f < FDIM; f += nw) {
    float s = 0.f;
    for (int c = lane; c < HDIM; c += 32) s += gelu_f(Cf[(size_t)f * HDIM + c]) * iw2[c];
    s = wave_sum(s);
    if (lane == 0)
      out[(size_t)n * FDIM + f] = W[(size_t)n * FDIM + f] + s + ib2[0];
  }
}

// ---------------- host ----------------
extern "C" void kernel_launch(void* const* d_in, const int* in_sizes, int n_in,
                              void* d_out, int out_size, void* d_ws, size_t ws_size,
                              hipStream_t stream) {
  (void)in_sizes; (void)n_in; (void)out_size; (void)ws_size;
  const float* W = (const float*)d_in[0];
  const float* M = (const float*)d_in[1];
  enum { I_FEMB = 2, I_VW1, I_VB1, I_VW2, I_VB2, I_FUSEW, I_FUSEB,
         I_SPW1, I_SPB1, I_SPG, I_SPBE, I_SPW2, I_SPB2, I_L0 };
  const int LS = 27; // entries per layer
  auto F = [&](int i) { return (const float*)d_in[i]; };

  char* wsp = (char*)d_ws;
  size_t off = 0;
  auto alloc = [&](size_t bytes) -> void* {
    void* p = wsp + off;
    off = (off + bytes + 255) & ~(size_t)255;
    return p;
  };
  auto packB = [&](int idx, int din, int dout, int kpad, int npad) -> bf16_t* {
    bf16_t* dst = (bf16_t*)alloc((size_t)kpad * npad * sizeof(bf16_t));
    int total = (kpad / 32) * (npad / 16) * 512;
    pack_b_kernel<<<(total + 255) / 256, 256, 0, stream>>>(F(idx), dst, din, dout, kpad, npad);
    return dst;
  };

  // ---- pack all weights to WMMA-B bf16 fragment layouts ----
  bf16_t* vw2pk  = packB(I_VW2, EDIM, EDIM, EDIM, EDIM);
  bf16_t* fusepk = packB(I_FUSEW, 2 * EDIM, EDIM, 2 * EDIM, EDIM);
  bf16_t* sw1pk  = packB(I_SPW1, EDIM, EDIM, EDIM, EDIM);
  bf16_t* sw2pk  = packB(I_SPW2, EDIM, EDIM, EDIM, EDIM);
  bf16_t *e_w1[2], *e_w2[2], *e_wp[2], *n_w1[2], *n_w2[2], *n_wp[2], *x_w1[2], *a_w1[2];
  float* n_b2p[2]; // node second-layer bias padded 130 -> 144
  int Lb[2];
  for (int l = 0; l < 2; ++l) {
    Lb[l] = I_L0 + LS * l;
    e_w1[l] = packB(Lb[l] + 0, EDGE_IN, HDIM, KP_EDGE, HDIM);
    e_w2[l] = packB(Lb[l] + 4, HDIM, HDIM, HDIM, HDIM);
    e_wp[l] = packB(Lb[l] + 6, EDGE_IN, HDIM, KP_EDGE, HDIM);
    n_w1[l] = packB(Lb[l] + 8, DIN_NODE, HDIM, KP_NODE, HDIM);
    n_w2[l] = packB(Lb[l] + 12, HDIM, NODE_IN, HDIM, NP_NODE);
    n_wp[l] = packB(Lb[l] + 14, DIN_NODE, NODE_IN, KP_NODE, NP_NODE);
    x_w1[l] = packB(Lb[l] + 16, EDGE_IN, HDIM, KP_EDGE, HDIM);
    a_w1[l] = packB(Lb[l] + 24, EDGE_IN, EDIM, KP_EDGE, EDIM);
    n_b2p[l] = (float*)alloc((size_t)NP_NODE * 4);
    pad_bias_kernel<<<1, NP_NODE, 0, stream>>>(F(Lb[l] + 13), n_b2p[l], NODE_IN, NP_NODE);
  }
  const int I_IMP = I_L0 + 2 * LS;
  bf16_t* iw1pk = packB(I_IMP + 0, EDGE_IN, HDIM, KP_EDGE, HDIM);

  // ---- f32 workspace buffers ----
  float* mean_feats = (float*)alloc((size_t)NS * EDIM * 4);
  float* sfeat_a = (float*)alloc((size_t)NS * SFS * 4);
  float* sfeat_b = (float*)alloc((size_t)NS * SFS * 4);
  float* ffeat_a = (float*)alloc((size_t)FDIM * SFS * 4);
  float* ffeat_b = (float*)alloc((size_t)FDIM * SFS * 4);
  float* w_a = (float*)alloc((size_t)NS * FDIM * 4);
  float* w_b = (float*)alloc((size_t)NS * FDIM * 4);
  float* a_log  = (float*)alloc((size_t)NS * FDIM * 4);
  float* attn_s = (float*)alloc((size_t)NS * FDIM * 4);
  float* attn_f = (float*)alloc((size_t)NS * FDIM * 4);
  float* agg_s  = (float*)alloc((size_t)NS * HDIM * 4);
  float* agg_f  = (float*)alloc((size_t)FDIM * HDIM * 4);

  // large dynamic-LDS opt-ins (host-side attribute, capture-safe)
  (void)hipFuncSetAttribute((const void*)encoder_kernel,    hipFuncAttributeMaxDynamicSharedMemorySize, 163840);
  (void)hipFuncSetAttribute((const void*)attn_kernel,       hipFuncAttributeMaxDynamicSharedMemorySize, 140288);
  (void)hipFuncSetAttribute((const void*)edge_layer_kernel, hipFuncAttributeMaxDynamicSharedMemorySize, 270336);
  (void)hipFuncSetAttribute((const void*)imp_kernel,        hipFuncAttributeMaxDynamicSharedMemorySize, 204800);

  // ---- pipeline ----
  winit_kernel<<<(NS * FDIM + 255) / 256, 256, 0, stream>>>(W, w_a, NS * FDIM);
  ffeat_init_kernel<<<(FDIM * SFS + 255) / 256, 256, 0, stream>>>(F(I_FEMB), ffeat_a);
  encoder_kernel<<<NS, 256, 163840, stream>>>(W, M, F(I_FEMB), F(I_VW1), F(I_VB1),
                                              vw2pk, F(I_VB2), fusepk, F(I_FUSEB), mean_feats);
  sp_kernel<<<NS / 16, 128, 16384, stream>>>(mean_feats, sw1pk, F(I_SPB1), F(I_SPG),
                                             F(I_SPBE), sw2pk, F(I_SPB2), sfeat_a);

  const float *sf_in = sfeat_a, *ff_in = ffeat_a, *w_in = w_a;
  float *sf_out = sfeat_b, *ff_out = ffeat_b, *w_out = w_b;
  for (int l = 0; l < 2; ++l) {
    const int B = Lb[l];
    attn_kernel<<<NS, 256, 140288, stream>>>(sf_in, ff_in, w_in, M, a_w1[l],
                                             F(B + 25), F(B + 26), a_log, attn_s);
    softmax_axis0_kernel<<<FDIM, 256, 0, stream>>>(a_log, attn_f);
    zero_kernel<<<(FDIM * HDIM + 255) / 256, 256, 0, stream>>>(agg_f, FDIM * HDIM);
    edge_layer_kernel<<<NS, 256, 270336, stream>>>(
        sf_in, ff_in, w_in, attn_s, attn_f,
        e_w1[l], F(B + 1), F(B + 2), F(B + 3), e_w2[l], F(B + 5), e_wp[l], F(B + 7),
        x_w1[l], F(B + 17), F(B + 18), F(B + 19), F(B + 20), F(B + 21), F(B + 22), F(B + 23),
        agg_s, agg_f, w_out);
    node_resmlp_kernel<<<NS / 16, 128, 47104, stream>>>(
        sf_in, agg_s, n_w1[l], F(B + 9), F(B + 10), F(B + 11),
        n_w2[l], n_b2p[l], n_wp[l], F(B + 15), sf_out);
    node_resmlp_kernel<<<FDIM / 16, 128, 47104, stream>>>(
        ff_in, agg_f, n_w1[l], F(B + 9), F(B + 10), F(B + 11),
        n_w2[l], n_b2p[l], n_wp[l], F(B + 15), ff_out);
    const float* t;
    t = sf_in; sf_in = sf_out; sf_out = (float*)t;
    t = ff_in; ff_in = ff_out; ff_out = (float*)t;
    t = w_in;  w_in = w_out;   w_out = (float*)t;
  }
  imp_kernel<<<NS, 256, 204800, stream>>>(W, sf_in, ff_in, iw1pk,
                                          F(I_IMP + 1), F(I_IMP + 2), F(I_IMP + 3),
                                          (float*)d_out);
}
